// Mamba3Block_71768903516836
// MI455X (gfx1250) — matrix operand order repaired and checked by reading, not compiled, not run on hardware
//
#include <hip/hip_runtime.h>

// ---------------------------------------------------------------------------
// Mamba3-style block for MI455X (gfx1250, wave32, WMMA).
// GEMM-shaped work -> v_wmma_f32_16x16x32_bf16 (fp32 accum).
// SSM: chunked scan (chunk=64), WMMA intra-chunk decay matmul, LDS carry,
//      async global->LDS staging (ASYNCcnt) for the chunk operand tiles.
// ---------------------------------------------------------------------------

typedef __attribute__((ext_vector_type(16))) __bf16 v16bf;
typedef __attribute__((ext_vector_type(8)))  __bf16 v8bf;
typedef __attribute__((ext_vector_type(8)))  float  v8f;

#define Bsz   2
#define Lseq  2048
#define DMdim 256
#define DTOT  1282
#define DZdim 512
#define DXdim 512
#define Hh    8
#define Pp    64
#define Nn    32
#define Rr    4
#define NP    2048          // N*P
#define CH    64            // scan chunk length
#define WSL   128           // width slice per workgroup in SSM kernel

static __device__ __forceinline__ __bf16 f2bf(float f) {
  union { float f; unsigned u; } a; a.f = f;
  unsigned u = a.u + 0x7FFFu + ((a.u >> 16) & 1u);
  union { unsigned short s; __bf16 b; } r; r.s = (unsigned short)(u >> 16);
  return r.b;
}
static __device__ __forceinline__ float bf2f(__bf16 b) {
  union { unsigned short s; __bf16 b; } a; a.b = b;
  union { unsigned u; float f; } r; r.u = ((unsigned)a.s) << 16;
  return r.f;
}
static __device__ __forceinline__ v16bf ldfrag(const __bf16* p0, const __bf16* p1) {
  union { v16bf v; v8bf h[2]; } u;
  u.h[0] = *(const v8bf*)p0;
  u.h[1] = *(const v8bf*)p1;
  return u.v;
}
// LDS byte offset of a __shared__ object (generic addr low 32 bits == LDS offset)
static __device__ __forceinline__ unsigned ldsoff(const void* p) {
  return (unsigned)(unsigned long long)p;
}
// async global->LDS copy of 8 bytes per lane (wave-level DMA, ASYNCcnt)
static __device__ __forceinline__ void async_g2l_b64(unsigned lds_byte, unsigned gbyte,
                                                     unsigned long long base) {
  asm volatile("global_load_async_to_lds_b64 %0, %1, %2"
               :
               : "v"(lds_byte), "v"(gbyte), "s"(base)
               : "memory");
}
static __device__ __forceinline__ void wait_async0() {
  asm volatile("s_wait_asynccnt 0x0" ::: "memory");
}

// ---------------------------------------------------------------------------
// 0) fp32 -> bf16 conversions (u, Wi (padded to 1296 rows), Wout)
// ---------------------------------------------------------------------------
__global__ void k_prep(const float* __restrict__ u, const float* __restrict__ Wi,
                       const float* __restrict__ Wout,
                       __bf16* __restrict__ ub, __bf16* __restrict__ wib,
                       __bf16* __restrict__ woutb) {
  for (size_t i = (size_t)blockIdx.x * 256 + threadIdx.x; i < (size_t)Bsz * Lseq * DMdim;
       i += (size_t)gridDim.x * 256) {
    ub[i] = f2bf(u[i]);
    if (i < (size_t)1296 * DMdim) wib[i] = (i < (size_t)DTOT * DMdim) ? f2bf(Wi[i]) : f2bf(0.f);
    if (i < (size_t)DMdim * DZdim) woutb[i] = f2bf(Wout[i]);
  }
}

// ---------------------------------------------------------------------------
// 1) proj = u @ Wi^T + bi    (M=4096, K=256, N=1282) -- WMMA bf16
// ---------------------------------------------------------------------------
__global__ __launch_bounds__(256) void k_proj(const __bf16* __restrict__ A,
                                              const __bf16* __restrict__ Bm,
                                              const float* __restrict__ bi,
                                              float* __restrict__ C) {
  const int lane = threadIdx.x & 31, wave = threadIdx.x >> 5;
  const int mt = blockIdx.x;
  const int nt = blockIdx.y * 8 + wave;
  if (nt >= 81) return;                       // wave-uniform: EXEC stays all-1s
  const int m0 = mt * 16, n0 = nt * 16;
  const __bf16* arow = A  + (size_t)(m0 + (lane & 15)) * DMdim + ((lane >> 4) * 8);
  const __bf16* brow = Bm + (size_t)(n0 + (lane & 15)) * DMdim + ((lane >> 4) * 8);
  v8f c = {0.f, 0.f, 0.f, 0.f, 0.f, 0.f, 0.f, 0.f};
  for (int kb = 0; kb < DMdim; kb += 32) {
    v16bf a  = ldfrag(arow + kb, arow + kb + 16);
    v16bf bb = ldfrag(brow + kb, brow + kb + 16);
    c = __builtin_amdgcn_wmma_f32_16x16x32_bf16(false, a, false, bb, (short)0, c, false, false);
  }
  const int col = n0 + (lane & 15);
  if (col < DTOT) {
    const float bv = bi[col];
    for (int i = 0; i < 8; i++) {
      const int row = m0 + i + ((lane >> 4) * 8);
      C[(size_t)row * DTOT + col] = c[i] + bv;
    }
  }
}

// ---------------------------------------------------------------------------
// 2) depthwise causal conv (k=4) + bias over xp slice of proj
// ---------------------------------------------------------------------------
__global__ void k_conv(const float* __restrict__ proj, const float* __restrict__ cw,
                       const float* __restrict__ cb, float* __restrict__ xconv) {
  size_t idx = (size_t)blockIdx.x * 256 + threadIdx.x;
  if (idx >= (size_t)Bsz * Lseq * DXdim) return;
  const int c = (int)(idx & 511);
  const size_t bl = idx >> 9;
  const int l = (int)(bl & (Lseq - 1));
  float acc = cb[c];
  for (int j = 0; j < 4; j++) {
    const int ls = l - 3 + j;
    if (ls >= 0) acc += cw[c * 4 + j] * proj[(bl - (size_t)(l - ls)) * DTOT + DZdim + c];
  }
  xconv[idx] = acc;
}

// ---------------------------------------------------------------------------
// 3) per-batch sequential scans: softplus(dt), angle cumsum, cs cumsum,
//    lam/alpha -> c1/c2 combine coefficients (head-independent since G=1)
// ---------------------------------------------------------------------------
__global__ void k_scan(const float* __restrict__ proj, const float* __restrict__ A_log,
                       const float* __restrict__ theta_log, float* __restrict__ angles,
                       float* __restrict__ cs, float* __restrict__ c1,
                       float* __restrict__ c2) {
  const int b = blockIdx.x, tid = threadIdx.x;
  const float A = -__expf(A_log[0]);
  const float th = (tid < 16) ? __expf(theta_log[tid]) : 0.f;
  float accA = 0.f, accC = 0.f;
  for (int l = 0; l < Lseq; l++) {
    const size_t row = (size_t)b * Lseq + l;
    const float dtr = proj[row * DTOT + 1280];
    const float dt = (dtr > 20.f) ? dtr : log1pf(__expf(dtr));
    if (tid < 16) { accA += dt * th; angles[row * 16 + tid] = accA; }
    if (tid == 16) {
      accC += dt * A; cs[row] = accC;
      const float lm = 1.f / (1.f + __expf(-proj[row * DTOT + 1281]));
      const float al = __expf(dt * A);
      c1[row] = lm * dt;
      c2[row] = (1.f - lm) * dt * al;
    }
  }
}

// ---------------------------------------------------------------------------
// 4) RMS-norm(B,C) + bias + RoPE  (heads share angles since G=1)
// ---------------------------------------------------------------------------
__global__ void k_bcn(const float* __restrict__ proj, const float* __restrict__ wB,
                      const float* __restrict__ wC, const float* __restrict__ biasB,
                      const float* __restrict__ biasC, const float* __restrict__ angles,
                      float* __restrict__ Brot, float* __restrict__ Crot) {
  __shared__ float Bn[128], Cn[128], rB[128], rC[128];
  const int tid = threadIdx.x;
  const size_t blk = blockIdx.x;
  const float* pr = proj + blk * DTOT;
  const float bv = pr[1024 + tid], cv = pr[1152 + tid];
  rB[tid] = bv * bv; rC[tid] = cv * cv;
  __syncthreads();
  for (int s = 64; s > 0; s >>= 1) {
    if (tid < s) { rB[tid] += rB[tid + s]; rC[tid] += rC[tid + s]; }
    __syncthreads();
  }
  const float rrB = rsqrtf(rB[0] / 128.f + 1e-5f);
  const float rrC = rsqrtf(rC[0] / 128.f + 1e-5f);
  Bn[tid] = bv * rrB * wB[tid] + biasB[tid];
  Cn[tid] = cv * rrC * wC[tid] + biasC[tid];
  __syncthreads();
  const int n = tid >> 2, n2 = n >> 1;
  const float ang = angles[blk * 16 + n2];
  const float c = __cosf(ang), s = __sinf(ang);
  float ob, oc;
  if ((n & 1) == 0) { ob = Bn[tid] * c - Bn[tid + 4] * s; oc = Cn[tid] * c - Cn[tid + 4] * s; }
  else              { ob = Bn[tid - 4] * s + Bn[tid] * c; oc = Cn[tid - 4] * s + Cn[tid] * c; }
  Brot[blk * 128 + tid] = ob;
  Crot[blk * 128 + tid] = oc;
}

// ---------------------------------------------------------------------------
// 5) xup = xh @ Wup^T, sig[n,p] = sum_r Brot[n,r]*xup[p,r]  -> bf16
// ---------------------------------------------------------------------------
__global__ __launch_bounds__(256) void k_xup(const float* __restrict__ xconv,
                                             const float* __restrict__ Wup,
                                             const float* __restrict__ Brot,
                                             __bf16* __restrict__ sig) {
  __shared__ float xh[64], xu[256], Bs[128];
  const int tid = threadIdx.x;
  const size_t blk = blockIdx.x;                      // b*L + l
  if (tid < 128) Bs[tid] = Brot[blk * 128 + tid];
  for (int h = 0; h < Hh; h++) {
    __syncthreads();
    if (tid < 64) xh[tid] = xconv[blk * DXdim + h * 64 + tid];
    __syncthreads();
    float acc = 0.f;
    for (int p = 0; p < 64; p++) acc += Wup[tid * 64 + p] * xh[p];
    xu[tid] = acc;
    __syncthreads();
    for (int idx = tid; idx < NP; idx += 256) {
      const int n = idx >> 6, p = idx & 63;
      float s = 0.f;
      for (int r = 0; r < 4; r++) s += Bs[n * 4 + r] * xu[p * 4 + r];
      sig[(blk * Hh + h) * NP + idx] = f2bf(s);
    }
  }
}

// ---------------------------------------------------------------------------
// 6) Chunked SSM scan: per (b,h,width-slice), for each 64-chunk:
//      async-DMA sig rows [c0-1 .. c0+63] into LDS (ASYNCcnt),
//      Lm[t][s]=exp(cs_t-cs_s) (s<=t), U[s][w]=c1*sig + c2*sig_prev,
//      H = Lm @ U (WMMA bf16) + exp(cs_t - cs_{c0-1}) * carry ; carry = H[63]
// ---------------------------------------------------------------------------
__global__ __launch_bounds__(256) void k_ssm(const __bf16* __restrict__ sig,
                                             const float* __restrict__ cs,
                                             const float* __restrict__ c1,
                                             const float* __restrict__ c2,
                                             __bf16* __restrict__ hout) {
  __shared__ alignas(32) __bf16 Lm[64 * 64];        // A: (t,s), ld=64
  __shared__ alignas(32) __bf16 Ut[WSL * 64];       // B transposed: (w,s), ld=64
  __shared__ alignas(32) __bf16 sraw[65 * WSL];     // raw sig rows gl=c0-1..c0+63
  __shared__ float carry[WSL];
  __shared__ float csL[64], dch[64], c1L[64], c2L[64];
  const int tid = threadIdx.x, lane = tid & 31, wave = tid >> 5;
  const int wsl = blockIdx.x, h = blockIdx.y, b = blockIdx.z;
  const int wbase = wsl * WSL;
  const size_t bl = (size_t)b * Lseq;
  const unsigned long long sigbase = (unsigned long long)(size_t)sig;
  if (tid < WSL) carry[tid] = 0.f;
  __syncthreads();
  for (int c0 = 0; c0 < Lseq; c0 += CH) {
    if (tid < 64) {
      const float cv = cs[bl + c0 + tid];
      csL[tid] = cv;
      const float csp = (c0 == 0) ? 0.f : cs[bl + c0 - 1];
      dch[tid] = __expf(cv - csp);
      c1L[tid] = c1[bl + c0 + tid];
      c2L[tid] = c2[bl + c0 + tid];
    }
    // --- async stage: 65 rows of 256B each, one wave-DMA per row ---
    for (int rr = wave; rr < 65; rr += 8) {          // wave-uniform trip count
      const int gl = c0 - 1 + rr;
      if (gl >= 0) {                                 // only false for rr==0,c0==0
        const unsigned ld = ldsoff(&sraw[rr * WSL + lane * 4]);
        const unsigned gb =
            (unsigned)((((bl + gl) * Hh + h) * (size_t)NP + wbase) * 2 + lane * 8);
        async_g2l_b64(ld, gb, sigbase);
      }
    }
    if (c0 == 0 && tid < WSL) sraw[tid] = f2bf(0.f); // "row -1" = zeros
    wait_async0();
    __syncthreads();
    // --- build decay matrix and transposed, combined U ---
    for (int idx = tid; idx < 64 * 64; idx += 256) {
      const int t = idx >> 6, s = idx & 63;
      Lm[idx] = f2bf((s <= t) ? __expf(csL[t] - csL[s]) : 0.f);
    }
    for (int idx = tid; idx < WSL * 64; idx += 256) {
      const int w = idx >> 6, s = idx & 63;
      const float cur = bf2f(sraw[(s + 1) * WSL + w]);
      const float prv = bf2f(sraw[s * WSL + w]);
      Ut[w * 64 + s] = f2bf(c1L[s] * cur + c2L[s] * prv);
    }
    __syncthreads();
    // --- H = Lm @ U via WMMA; add carry; emit; harvest row 63 ---
    float cnew = 0.f; int cncol = -1;
    for (int k = 0; k < 4; k++) {
      const int id = wave + 8 * k;                   // 32 tiles over 8 waves
      const int mt = id >> 3, nt = id & 7;
      const int m0 = mt * 16, n0 = nt * 16;
      const __bf16* arow = Lm + (m0 + (lane & 15)) * 64 + ((lane >> 4) * 8);
      const __bf16* brow = Ut + (n0 + (lane & 15)) * 64 + ((lane >> 4) * 8);
      v8f c = {0.f, 0.f, 0.f, 0.f, 0.f, 0.f, 0.f, 0.f};
      for (int kb = 0; kb < 64; kb += 32) {
        v16bf a  = ldfrag(arow + kb, arow + kb + 16);
        v16bf bb = ldfrag(brow + kb, brow + kb + 16);
        c = __builtin_amdgcn_wmma_f32_16x16x32_bf16(false, a, false, bb, (short)0, c, false, false);
      }
      const int col = n0 + (lane & 15);
      const float cw = carry[col];
      for (int i = 0; i < 8; i++) {
        const int rl = m0 + i + ((lane >> 4) * 8);
        const float hv = c[i] + dch[rl] * cw;
        hout[(((size_t)(b * Hh + h) * Lseq + (c0 + rl)) * NP) + wbase + col] = f2bf(hv);
        if (rl == 63) { cnew = hv; cncol = col; }
      }
    }
    __syncthreads();
    if (cncol >= 0) carry[cncol] = cnew;
    __syncthreads();
  }
}

// ---------------------------------------------------------------------------
// 7) y = C-contraction, yd = Wdn @ y, + D*xconv, SiLU(z) gate -> bf16
//    h row staged to LDS via async global->LDS DMA.
// ---------------------------------------------------------------------------
__global__ __launch_bounds__(256) void k_y(const __bf16* __restrict__ hbuf,
                                           const float* __restrict__ Crot,
                                           const float* __restrict__ Wdn,
                                           const float* __restrict__ xconv,
                                           const float* __restrict__ proj,
                                           const float* __restrict__ Dvec,
                                           __bf16* __restrict__ ygate) {
  __shared__ alignas(32) __bf16 hrow[NP];
  __shared__ float Cs[128], yfl[256];
  const int tid = threadIdx.x, lane = tid & 31, wave = tid >> 5;
  const size_t blk = blockIdx.x;
  const int b = (int)(blk >> 11), l = (int)(blk & (Lseq - 1));
  const unsigned long long hbase = (unsigned long long)(size_t)hbuf;
  if (tid < 128) Cs[tid] = Crot[blk * 128 + tid];
  for (int h = 0; h < Hh; h++) {
    __syncthreads();
    const size_t rowbytes = ((size_t)(b * Hh + h) * Lseq + l) * NP * 2;
    for (int ii = wave * 2; ii < wave * 2 + 2; ii++) {   // 16 x 256B DMA ops
      const unsigned ld = ldsoff(&hrow[ii * 128 + lane * 4]);
      const unsigned gb = (unsigned)(rowbytes + (size_t)ii * 256 + lane * 8);
      async_g2l_b64(ld, gb, hbase);
    }
    wait_async0();
    __syncthreads();
    {
      const int p = tid >> 2, r = tid & 3;
      float acc = 0.f;
      for (int n = 0; n < 32; n++) acc += bf2f(hrow[n * 64 + p]) * Cs[n * 4 + r];
      yfl[tid] = acc;                       // q = p*R + r
    }
    __syncthreads();
    if (tid < 64) {
      float acc = 0.f;
      for (int q = 0; q < 256; q++) acc += Wdn[tid * 256 + q] * yfl[q];
      const int col = h * 64 + tid;
      float yf = acc + xconv[blk * DXdim + col] * Dvec[h];
      const float z = proj[blk * DTOT + col];
      yf *= z * (1.f / (1.f + __expf(-z)));
      ygate[blk * DZdim + col] = f2bf(yf);
    }
  }
}

// ---------------------------------------------------------------------------
// 8) out = ygate @ Wout^T  (M=4096, K=512, N=256) -- WMMA bf16, fp32 out
// ---------------------------------------------------------------------------
__global__ __launch_bounds__(256) void k_out(const __bf16* __restrict__ A,
                                             const __bf16* __restrict__ Bm,
                                             float* __restrict__ C) {
  const int lane = threadIdx.x & 31, wave = threadIdx.x >> 5;
  const int mt = blockIdx.x;
  const int nt = blockIdx.y * 8 + wave;
  const int m0 = mt * 16, n0 = nt * 16;
  const __bf16* arow = A  + (size_t)(m0 + (lane & 15)) * DZdim + ((lane >> 4) * 8);
  const __bf16* brow = Bm + (size_t)(n0 + (lane & 15)) * DZdim + ((lane >> 4) * 8);
  v8f c = {0.f, 0.f, 0.f, 0.f, 0.f, 0.f, 0.f, 0.f};
  for (int kb = 0; kb < DZdim; kb += 32) {
    v16bf a  = ldfrag(arow + kb, arow + kb + 16);
    v16bf bb = ldfrag(brow + kb, brow + kb + 16);
    c = __builtin_amdgcn_wmma_f32_16x16x32_bf16(false, a, false, bb, (short)0, c, false, false);
  }
  const int col = n0 + (lane & 15);
  for (int i = 0; i < 8; i++) {
    const int row = m0 + i + ((lane >> 4) * 8);
    C[(size_t)row * DMdim + col] = c[i];
  }
}

// ---------------------------------------------------------------------------
extern "C" void kernel_launch(void* const* d_in, const int* in_sizes, int n_in,
                              void* d_out, int out_size, void* d_ws, size_t ws_size,
                              hipStream_t stream) {
  (void)in_sizes; (void)n_in; (void)out_size; (void)ws_size;
  const float* u        = (const float*)d_in[0];
  const float* Wi       = (const float*)d_in[1];
  const float* bi       = (const float*)d_in[2];
  const float* conv_w   = (const float*)d_in[3];
  const float* conv_b   = (const float*)d_in[4];
  const float* Wup      = (const float*)d_in[5];
  const float* Wdn      = (const float*)d_in[6];
  const float* A_log    = (const float*)d_in[7];
  const float* theta_lg = (const float*)d_in[8];
  const float* Dv       = (const float*)d_in[9];
  const float* wB       = (const float*)d_in[10];
  const float* wC       = (const float*)d_in[11];
  const float* bias_B   = (const float*)d_in[12];
  const float* bias_C   = (const float*)d_in[13];
  const float* Wout     = (const float*)d_in[14];

  char* w = (char*)d_ws;
  auto alloc = [&](size_t bytes) -> char* {
    char* p = w; w += (bytes + 255) & ~(size_t)255; return p;
  };
  const size_t BL = (size_t)Bsz * Lseq;                    // 4096
  __bf16* ub     = (__bf16*)alloc(BL * DMdim * 2);
  __bf16* wib    = (__bf16*)alloc((size_t)1296 * DMdim * 2);
  __bf16* woutb  = (__bf16*)alloc((size_t)DMdim * DZdim * 2);
  float*  proj   = (float*)alloc(BL * DTOT * 4);
  float*  xconv  = (float*)alloc(BL * DXdim * 4);
  float*  angles = (float*)alloc(BL * 16 * 4);
  float*  cs     = (float*)alloc(BL * 4);
  float*  c1     = (float*)alloc(BL * 4);
  float*  c2     = (float*)alloc(BL * 4);
  float*  Brot   = (float*)alloc(BL * 128 * 4);
  float*  Crot   = (float*)alloc(BL * 128 * 4);
  __bf16* sig    = (__bf16*)alloc(BL * Hh * NP * 2);       // 134 MB
  __bf16* hbuf   = (__bf16*)alloc(BL * Hh * NP * 2);       // 134 MB
  __bf16* ygate  = (__bf16*)alloc(BL * DZdim * 2);

  k_prep<<<4096, 256, 0, stream>>>(u, Wi, Wout, ub, wib, woutb);
  k_proj<<<dim3(256, 11), 256, 0, stream>>>(ub, wib, bi, proj);
  k_conv<<<(unsigned)((BL * DXdim) / 256), 256, 0, stream>>>(proj, conv_w, conv_b, xconv);
  k_scan<<<Bsz, 32, 0, stream>>>(proj, A_log, theta_lg, angles, cs, c1, c2);
  k_bcn<<<(unsigned)BL, 128, 0, stream>>>(proj, wB, wC, bias_B, bias_C, angles, Brot, Crot);
  k_xup<<<(unsigned)BL, 256, 0, stream>>>(xconv, Wup, Brot, sig);
  k_ssm<<<dim3(NP / WSL, Hh, Bsz), 256, 0, stream>>>(sig, cs, c1, c2, hbuf);
  k_y<<<(unsigned)BL, 256, 0, stream>>>(hbuf, Crot, Wdn, xconv, proj, Dv, ygate);
  k_out<<<dim3(256, 2), 256, 0, stream>>>(ygate, woutb, (float*)d_out);
}
